// Attention_1915555414563
// MI455X (gfx1250) — compile-verified, matrix operand order checked by reading, not run on hardware
//
#include <hip/hip_runtime.h>

// Problem constants (match reference)
#define B_    64
#define N_    512
#define DIM_  256
#define H_    8
#define DH_   64
#define INNER_ 512
#define SCALE_ 0.125f   // 64^-0.5

typedef __attribute__((ext_vector_type(16))) __bf16 bf16x16;
typedef __attribute__((ext_vector_type(8)))  __bf16 bf16x8;
typedef __attribute__((ext_vector_type(8)))  float  f32x8;

// Load a 16-element bf16 A/B fragment slice for v_wmma_f32_16x16x32_bf16.
// Per ISA 7.12.2: for a given lane, VGPRs 0..3 hold K = base..base+7 (pairs),
// VGPRs 4..7 hold K = base+16..base+23, where base = k0 + 8*(lane>=16).
static __device__ __forceinline__ bf16x16 load_frag(const __bf16* p) {
  bf16x8 lo = *(const bf16x8*)(p);
  bf16x8 hi = *(const bf16x8*)(p + 16);
  return __builtin_shufflevector(lo, hi,
      0,1,2,3,4,5,6,7,8,9,10,11,12,13,14,15);
}

static __device__ __forceinline__ f32x8 wmma_bf16(bf16x16 a, bf16x16 b, f32x8 c) {
  return __builtin_amdgcn_wmma_f32_16x16x32_bf16(
      /*neg_a=*/false, a, /*neg_b=*/false, b,
      /*c_mod=*/(short)0, c, /*reuse_a=*/false, /*reuse_b=*/false);
}

// ---------------------------------------------------------------- prep kernels
__global__ void convert_x_kernel(const float* __restrict__ src,
                                 __bf16* __restrict__ dst, int n) {
  int i = blockIdx.x * blockDim.x + threadIdx.x;
  if (i < n) dst[i] = (__bf16)src[i];
}

// dst[o*K + i] = src[i*ncols_full + col0 + o]   (weight -> [out][in] bf16)
__global__ void transpose_w_kernel(const float* __restrict__ src,
                                   int ncols_full, int col0, int ncols, int K,
                                   __bf16* __restrict__ dst) {
  int idx = blockIdx.x * blockDim.x + threadIdx.x;
  if (idx >= ncols * K) return;
  int o = idx / K, i = idx - o * K;
  dst[(size_t)o * K + i] = (__bf16)src[(size_t)i * ncols_full + col0 + o];
}

// ---------------------------------------------------------- blocked tile GEMM
// One wave computes a 32(M) x 64(N) block of A[M,K] @ Bt[N,K]^T:
// per K-step it loads 2 A fragments + 4 B fragments (12 b128 loads) and
// issues 8 v_wmma_f32_16x16x32_bf16 -> 1.5 loads/WMMA (vs 4.0 for 16x16).
// Stores per `mode`:
//   0: Q  (scale, [b,h,n,d] bf16)      1: K  ([b,h,n,d] bf16)
//   2: V  ([b,h,d,n] bf16, transposed) 3: G  ([m, 512] bf16, +bias)
//   4: final output ([m, 256] f32, +bias)
__global__ __launch_bounds__(32)
void gemm_tile_kernel(const __bf16* __restrict__ A, const __bf16* __restrict__ Bt,
                      int K, int mode, __bf16* __restrict__ obf,
                      float* __restrict__ ofp, const float* __restrict__ bias) {
  const int lane = threadIdx.x;
  const int hi = lane >> 4;      // which half-wave
  const int ln = lane & 15;
  const int m0 = blockIdx.y * 32;
  const int n0 = blockIdx.x * 64;

  const __bf16* arow0 = A  + (size_t)(m0 + ln) * K;        // rows m0..m0+15
  const __bf16* arow1 = A  + (size_t)(m0 + 16 + ln) * K;   // rows m0+16..m0+31
  const __bf16* brow0 = Bt + (size_t)(n0 + ln) * K;        // cols n0+f*16..
  const __bf16* brow1 = Bt + (size_t)(n0 + 16 + ln) * K;
  const __bf16* brow2 = Bt + (size_t)(n0 + 32 + ln) * K;
  const __bf16* brow3 = Bt + (size_t)(n0 + 48 + ln) * K;

  f32x8 acc[2][4] = {};
  for (int k0 = 0; k0 < K; k0 += 32) {
    const int kb = k0 + (hi << 3);
    __builtin_prefetch(arow0 + kb + 64, 0, 0);   // global_prefetch_b8
    __builtin_prefetch(brow0 + kb + 64, 0, 0);
    bf16x16 a0 = load_frag(arow0 + kb);
    bf16x16 a1 = load_frag(arow1 + kb);
    bf16x16 b0 = load_frag(brow0 + kb);
    bf16x16 b1 = load_frag(brow1 + kb);
    bf16x16 b2 = load_frag(brow2 + kb);
    bf16x16 b3 = load_frag(brow3 + kb);
    acc[0][0] = wmma_bf16(a0, b0, acc[0][0]);
    acc[0][1] = wmma_bf16(a0, b1, acc[0][1]);
    acc[0][2] = wmma_bf16(a0, b2, acc[0][2]);
    acc[0][3] = wmma_bf16(a0, b3, acc[0][3]);
    acc[1][0] = wmma_bf16(a1, b0, acc[1][0]);
    acc[1][1] = wmma_bf16(a1, b1, acc[1][1]);
    acc[1][2] = wmma_bf16(a1, b2, acc[1][2]);
    acc[1][3] = wmma_bf16(a1, b3, acc[1][3]);
  }

#pragma unroll
  for (int mi = 0; mi < 2; ++mi) {
#pragma unroll
    for (int f = 0; f < 4; ++f) {
#pragma unroll
      for (int r = 0; r < 8; ++r) {
        // C/D layout: VGPR r -> row +r (lanes 0-15) or +r+8 (lanes 16-31)
        const int m = m0 + mi * 16 + r + (hi << 3);
        const int n = n0 + f * 16 + ln;
        float v = acc[mi][f][r];
        if (mode <= 2) {               // q/k/v scatter into per-head layouts
          const int b = m >> 9;        // m / 512
          const int i = m & 511;
          const int h = n >> 6;        // n / 64
          const int d = n & 63;
          const size_t bh = (size_t)b * H_ + h;
          if (mode == 0)      obf[(bh * N_ + i) * DH_ + d] = (__bf16)(v * SCALE_);
          else if (mode == 1) obf[(bh * N_ + i) * DH_ + d] = (__bf16)v;
          else                obf[(bh * DH_ + d) * N_ + i] = (__bf16)v;  // V^T
        } else if (mode == 3) {        // gates = x@Wg + bg
          obf[(size_t)m * INNER_ + n] = (__bf16)(v + bias[n]);
        } else {                       // final: out = AO@Wo + bo (f32)
          ofp[(size_t)m * DIM_ + n] = v + bias[n];
        }
      }
    }
  }
}

// -------------------------------------------------------- flash attention core
// One wave per (b, h, 16-row i-tile). Online softmax over j in steps of 32.
__global__ __launch_bounds__(32)
void attn_kernel(const __bf16* __restrict__ qh, const __bf16* __restrict__ kh,
                 const __bf16* __restrict__ vt, const __bf16* __restrict__ g,
                 const float* __restrict__ bias,
                 const unsigned char* __restrict__ mask,
                 __bf16* __restrict__ ao) {
  // P-tile staging buffer: 16 rows x 32 cols, padded to 40 (80B row stride,
  // keeps 16B alignment and spreads banks).
  __shared__ __align__(16) __bf16 pbuf[16 * 40];

  const int lane = threadIdx.x;
  const int hi = lane >> 4;
  const int ln = lane & 15;
  const int i0 = blockIdx.x * 16;
  const int h  = blockIdx.y;
  const int b  = blockIdx.z;
  const size_t bh = (size_t)b * H_ + h;

  // Q fragments for K=0..31 and K=32..63 (loaded once; Q is pre-scaled)
  const __bf16* qrow = qh + (bh * N_ + i0 + ln) * DH_;
  const bf16x16 qf0 = load_frag(qrow + (hi << 3));
  const bf16x16 qf1 = load_frag(qrow + 32 + (hi << 3));

  unsigned char mrow[8];
#pragma unroll
  for (int r = 0; r < 8; ++r)
    mrow[r] = mask[(size_t)b * N_ + i0 + r + (hi << 3)];

  float rowm[8], rows[8];
  f32x8 acc[4] = {};
#pragma unroll
  for (int r = 0; r < 8; ++r) { rowm[r] = -3.0e38f; rows[r] = 0.0f; }

  for (int j0 = 0; j0 < N_; j0 += 32) {
    // ---- scores: two 16-col tiles, K=64 in two chunks -> 4 WMMAs
    const __bf16* krow0 = kh + (bh * N_ + j0 + ln) * DH_;
    const __bf16* krow1 = krow0 + (size_t)16 * DH_;
    f32x8 d0 = {}, d1 = {};
    d0 = wmma_bf16(qf0, load_frag(krow0 + (hi << 3)), d0);
    d0 = wmma_bf16(qf1, load_frag(krow0 + 32 + (hi << 3)), d0);
    d1 = wmma_bf16(qf0, load_frag(krow1 + (hi << 3)), d1);
    d1 = wmma_bf16(qf1, load_frag(krow1 + 32 + (hi << 3)), d1);

    const unsigned char mc0 = mask[(size_t)b * N_ + j0 + ln];
    const unsigned char mc1 = mask[(size_t)b * N_ + j0 + 16 + ln];

    float fac[8];
#pragma unroll
    for (int r = 0; r < 8; ++r) {
      const int ri = r + (hi << 3);
      const float* bp = bias + (((size_t)h * N_ + i0 + ri) * N_) + j0;
      float v0 = d0[r] + bp[ln];
      float v1 = d1[r] + bp[16 + ln];
      if (!(mrow[r] && mc0)) v0 = -3.0e38f;
      if (!(mrow[r] && mc1)) v1 = -3.0e38f;
      // row max over the 16 lanes holding this row (stay within half-wave)
      float mx = fmaxf(v0, v1);
#pragma unroll
      for (int s = 8; s; s >>= 1) mx = fmaxf(mx, __shfl_xor(mx, s, 32));
      const float nm = fmaxf(rowm[r], mx);
      fac[r] = __expf(rowm[r] - nm);
      const float p0 = __expf(v0 - nm);
      const float p1 = __expf(v1 - nm);
      float ts = p0 + p1;
#pragma unroll
      for (int s = 8; s; s >>= 1) ts += __shfl_xor(ts, s, 32);
      rows[r] = rows[r] * fac[r] + ts;
      rowm[r] = nm;
      pbuf[ri * 40 + ln]      = (__bf16)p0;
      pbuf[ri * 40 + 16 + ln] = (__bf16)p1;
    }
#pragma unroll
    for (int f = 0; f < 4; ++f)
#pragma unroll
      for (int r = 0; r < 8; ++r) acc[f][r] *= fac[r];

    __syncthreads();   // LDS visibility of P-tile across lanes

    // ---- P @ V : P is the A fragment (K = 32 j-columns), V^T rows contiguous
    const bf16x16 pf = load_frag(&pbuf[ln * 40 + (hi << 3)]);
#pragma unroll
    for (int f = 0; f < 4; ++f) {
      const __bf16* vr = vt + (bh * DH_ + f * 16 + ln) * N_ + j0 + (hi << 3);
      acc[f] = wmma_bf16(pf, load_frag(vr), acc[f]);
    }
    __syncthreads();   // safe to overwrite pbuf next iteration
  }

  // ---- finalize: 1/rowsum, apply gate, store gated bf16 activations
#pragma unroll
  for (int r = 0; r < 8; ++r) {
    const int ri = r + (hi << 3);
    const size_t m = (size_t)b * N_ + i0 + ri;
    const float inv = 1.0f / rows[r];
#pragma unroll
    for (int f = 0; f < 4; ++f) {
      const int col = h * DH_ + f * 16 + ln;
      const float gate = (float)g[m * INNER_ + col];
      ao[m * INNER_ + col] = (__bf16)(acc[f][r] * inv * gate);
    }
  }
}

// ------------------------------------------------------------------- launcher
extern "C" void kernel_launch(void* const* d_in, const int* in_sizes, int n_in,
                              void* d_out, int out_size, void* d_ws, size_t ws_size,
                              hipStream_t stream) {
  const float*         x    = (const float*)d_in[0];
  const unsigned char* mask = (const unsigned char*)d_in[1];   // jax bool (1B)
  const float*         bias = (const float*)d_in[2];
  const float*         Wq   = (const float*)d_in[3];
  const float*         Wkv  = (const float*)d_in[4];
  const float*         Wo   = (const float*)d_in[5];
  const float*         bo   = (const float*)d_in[6];
  const float*         Wg   = (const float*)d_in[7];
  const float*         bg   = (const float*)d_in[8];
  float* out = (float*)d_out;

  const size_t M = (size_t)B_ * N_;            // 32768 rows
  __bf16* ws = (__bf16*)d_ws;
  __bf16* xh  = ws;                  ws += M * DIM_;          // x in bf16
  __bf16* qh  = ws;                  ws += M * INNER_;        // [b,h,n,d]
  __bf16* kh  = ws;                  ws += M * INNER_;        // [b,h,n,d]
  __bf16* vt  = ws;                  ws += M * INNER_;        // [b,h,d,n]
  __bf16* gg  = ws;                  ws += M * INNER_;        // gates bf16
  __bf16* ao  = ws;                  ws += M * INNER_;        // gated attn out
  __bf16* wqt = ws;                  ws += (size_t)INNER_ * DIM_;
  __bf16* wkt = ws;                  ws += (size_t)INNER_ * DIM_;
  __bf16* wvt = ws;                  ws += (size_t)INNER_ * DIM_;
  __bf16* wgt = ws;                  ws += (size_t)INNER_ * DIM_;
  __bf16* wot = ws;                  ws += (size_t)DIM_ * INNER_;

  // 1) conversions / weight transposes (bf16, [out][in])
  {
    int n = (int)(M * DIM_);
    convert_x_kernel<<<(n + 255) / 256, 256, 0, stream>>>(x, xh, n);
    int nw = INNER_ * DIM_;   // 131072 elements each
    transpose_w_kernel<<<(nw + 255) / 256, 256, 0, stream>>>(Wq,  INNER_,     0, INNER_, DIM_, wqt);
    transpose_w_kernel<<<(nw + 255) / 256, 256, 0, stream>>>(Wkv, 2 * INNER_, 0, INNER_, DIM_, wkt);
    transpose_w_kernel<<<(nw + 255) / 256, 256, 0, stream>>>(Wkv, 2 * INNER_, INNER_, INNER_, DIM_, wvt);
    transpose_w_kernel<<<(nw + 255) / 256, 256, 0, stream>>>(Wg,  INNER_,     0, INNER_, DIM_, wgt);
    transpose_w_kernel<<<(nw + 255) / 256, 256, 0, stream>>>(Wo,  DIM_,       0, DIM_,  INNER_, wot);
  }

  const dim3 blk(32);
  // 32x64 output block per wave
  const dim3 gP(INNER_ / 64, (unsigned)(M / 32));       // 8 x 1024 blocks

  // 2) projections (WMMA bf16)
  gemm_tile_kernel<<<gP, blk, 0, stream>>>(xh, wqt, DIM_, 0, qh, nullptr, nullptr); // Q (scaled)
  gemm_tile_kernel<<<gP, blk, 0, stream>>>(xh, wkt, DIM_, 1, kh, nullptr, nullptr); // K
  gemm_tile_kernel<<<gP, blk, 0, stream>>>(xh, wvt, DIM_, 2, vt, nullptr, nullptr); // V^T
  gemm_tile_kernel<<<gP, blk, 0, stream>>>(xh, wgt, DIM_, 3, gg, nullptr, bg);      // gates

  // 3) flash attention + gating (WMMA bf16, fp32 softmax)
  attn_kernel<<<dim3(N_ / 16, H_, B_), blk, 0, stream>>>(qh, kh, vt, gg, bias, mask, ao);

  // 4) output projection: out = AO @ Wo + bo (fp32 result)
  gemm_tile_kernel<<<dim3(DIM_ / 64, (unsigned)(M / 32)), blk, 0, stream>>>(
      ao, wot, INNER_, 4, nullptr, out, bo);
}
